// DglGATModel_57174604644835
// MI455X (gfx1250) — matrix operand order repaired and checked by reading, not compiled, not run on hardware
//
#include <hip/hip_runtime.h>
#include <hip/hip_bf16.h>
#include <math.h>
#include <stdint.h>

// ---------------------------------------------------------------------------
// CDNA5 (gfx1250) GAT pipeline.
// Dense projections: v_wmma_f32_16x16x32_f16, 4 waves/block, B tile staged to
// LDS via global_load_async_to_lds_b32 (ASYNCcnt) and read back via ds_load.
// Sparse attention: L2-resident atomics (feat arrays are 51.2MB < 192MB L2).
// ---------------------------------------------------------------------------

typedef __attribute__((ext_vector_type(16))) _Float16 v16h;
typedef __attribute__((ext_vector_type(8)))  float    v8f;

#define N_NODES 50000
#define N_EDGES 800000
#define FDIM    256
#define SLOPE   0.2f

#define GEMM_TPB 128   // 4 waves -> 64x16 C strip per block

// ---------------------------------------------------------------------------
// GEMM: C[M,N] = A[M,K] * B[K,N], fp32 in/out, f16 WMMA inside.
// Block = 4 waves; wave w owns 16x16 tile at rows (blockIdx.x*4+w)*16.
// Per k-step (32):
//   - 128 threads async-copy the 32x16 fp32 B tile into LDS (pad stride 17:
//     lanes 0-15 and 16-31 hit bank sets shifted by 8, overlapping, on readback).
//   - each wave loads its A rows as 4x global_load_b128 (contiguous, aligned).
// Out-of-range A rows are CLAMPED (not predicated): garbage only reaches
// D rows >= M, which the guarded store drops -> loads stay unconditional.
// Per-lane operand layouts follow CDNA5 ISA 7.12.2:
//   A 16x32 f16 : lane L holds row L%16; halves i -> k = (i&7) + (i>=8?16:0)
//                 + (L>=16 ? 8 : 0)
//   B 32x16 f16 : lane L holds col L%16; halves i -> same k as A
//   C/D 16x16 f32: VGPR v -> row v + (L>=16 ? 8 : 0), col L%16
// ---------------------------------------------------------------------------
__global__ void gat_gemm_wmma(const float* __restrict__ A,
                              const float* __restrict__ B,
                              float* __restrict__ C,
                              int M, int K, int N) {
    __shared__ float lds_b[32 * 17];        // 32 K-rows x 16 cols, pad->17

    const int tid  = threadIdx.x;
    const int wave = tid >> 5;
    const int lane = tid & 31;
    const int tm   = (blockIdx.x * 4 + wave) * 16;
    const int tn   = blockIdx.y * 16;
    const int r    = lane & 15;
    const int hi   = lane >> 4;             // 0 or 1

    int arow = tm + r;
    if (arow >= M) arow = M - 1;            // clamp: safe, see note above
    const int bcol = tn + r;

    const float* Arow = A + (size_t)arow * K + hi * 8;

    v8f acc = {};
    for (int k0 = 0; k0 < K; k0 += 32) {
        // ---- stage B tile: global -> LDS, async (ASYNCcnt) ----
        {
            int idx = tid;                   // 512 elements / 128 threads
            #pragma unroll
            for (int j = 0; j < 4; ++j, idx += GEMM_TPB) {
                int kr = idx >> 4;           // 0..31
                int c  = idx & 15;           // 0..15
                uint32_t ldsOff = (uint32_t)(uintptr_t)&lds_b[kr * 17 + c];
                const float* gp = B + (size_t)(k0 + kr) * N + tn + c;
                asm volatile("global_load_async_to_lds_b32 %0, %1, off"
                             :: "v"(ldsOff), "v"(gp) : "memory");
            }
            asm volatile("s_wait_asynccnt 0" ::: "memory");
        }
        __syncthreads();

        // ---- A: two contiguous 8-float runs -> 4x global_load_b128 ----
        const float4* ap0 = (const float4*)(Arow + k0);        // k0   + hi*8 ..
        const float4* ap1 = (const float4*)(Arow + k0 + 16);   // k0+16+ hi*8 ..
        float4 a0 = ap0[0], a1 = ap0[1];
        float4 a2 = ap1[0], a3 = ap1[1];

        v16h a;
        a[0]  = (_Float16)a0.x; a[1]  = (_Float16)a0.y;
        a[2]  = (_Float16)a0.z; a[3]  = (_Float16)a0.w;
        a[4]  = (_Float16)a1.x; a[5]  = (_Float16)a1.y;
        a[6]  = (_Float16)a1.z; a[7]  = (_Float16)a1.w;
        a[8]  = (_Float16)a2.x; a[9]  = (_Float16)a2.y;
        a[10] = (_Float16)a2.z; a[11] = (_Float16)a2.w;
        a[12] = (_Float16)a3.x; a[13] = (_Float16)a3.y;
        a[14] = (_Float16)a3.z; a[15] = (_Float16)a3.w;

        // ---- B: read f16 operand from LDS (stride 17) ----
        v16h b;
        #pragma unroll
        for (int i = 0; i < 16; ++i)
            b[i] = (_Float16)lds_b[((i & 7) + (i >= 8 ? 16 : 0) + hi * 8) * 17 + r];

        if (k0 + 32 < K)                     // uniform branch, EXEC stays full
            __builtin_prefetch(Arow + k0 + 32, 0, 0);   // global_prefetch_b8

        acc = __builtin_amdgcn_wmma_f32_16x16x32_f16(
                  false, a, false, b, (short)0, acc, false, false);

        __syncthreads();                     // protect lds_b before next fill
    }

    #pragma unroll
    for (int v = 0; v < 8; ++v) {
        int row = tm + v + hi * 8;
        if (row < M) C[(size_t)row * N + bcol] = acc[v];
    }
}

// el[n,h] = <feat[n,h,:], attn_l[h,:]> ; er likewise
__global__ void gat_node_attn(const float* __restrict__ feat,
                              const float* __restrict__ al,
                              const float* __restrict__ ar,
                              float* __restrict__ el, float* __restrict__ er,
                              int N, int H, int D) {
    int idx = blockIdx.x * blockDim.x + threadIdx.x;
    if (idx >= N * H) return;
    int n = idx / H, h = idx - n * H;
    const float* f = feat + (size_t)n * H * D + (size_t)h * D;
    const float* l = al + h * D;
    const float* q = ar + h * D;
    float sl = 0.f, sr = 0.f;
    for (int d = 0; d < D; ++d) { float v = f[d]; sl += v * l[d]; sr += v * q[d]; }
    el[idx] = sl; er[idx] = sr;
}

__global__ void fill_f32(float* __restrict__ p, float v, size_t n) {
    size_t i = (size_t)blockIdx.x * blockDim.x + threadIdx.x;
    if (i < n) p[i] = v;
}

__device__ __forceinline__ float leaky(float v) {
    return v > 0.f ? v : SLOPE * v;
}

// Signed/unsigned bit-pattern float atomic max (emax initialized to -inf).
__device__ __forceinline__ void atomicMaxF(float* addr, float val) {
    if (val >= 0.f) atomicMax((int*)addr, __float_as_int(val));
    else            atomicMin((unsigned int*)addr, __float_as_uint(val));
}

__global__ void gat_edge_max(const int* __restrict__ src, const int* __restrict__ dst,
                             const float* __restrict__ el, const float* __restrict__ er,
                             float* __restrict__ emax, int E, int H) {
    int idx = blockIdx.x * blockDim.x + threadIdx.x;
    if (idx >= E * H) return;
    int e = idx / H, h = idx - e * H;
    int s = src[e], d = dst[e];
    float v = leaky(el[s * H + h] + er[d * H + h]);
    atomicMaxF(&emax[d * H + h], v);
}

__global__ void gat_edge_sum(const int* __restrict__ src, const int* __restrict__ dst,
                             const float* __restrict__ el, const float* __restrict__ er,
                             const float* __restrict__ emax, float* __restrict__ denom,
                             int E, int H) {
    int idx = blockIdx.x * blockDim.x + threadIdx.x;
    if (idx >= E * H) return;
    int e = idx / H, h = idx - e * H;
    int s = src[e], d = dst[e];
    float v  = leaky(el[s * H + h] + er[d * H + h]);
    float ee = expf(v - emax[d * H + h]);
    atomicAdd(&denom[d * H + h], ee);
}

// rst[dst,h,d] += a(e,h) * feat[src,h,d]; a recomputed from node-level stats
// (el/er/emax/denom: 800KB, L2-resident -> no ExH edge buffers needed).
__global__ void gat_aggregate(const int* __restrict__ src, const int* __restrict__ dst,
                              const float* __restrict__ feat,
                              const float* __restrict__ el, const float* __restrict__ er,
                              const float* __restrict__ emax, const float* __restrict__ denom,
                              float* __restrict__ rst,
                              long total, int H, int D) {
    size_t idx = (size_t)blockIdx.x * blockDim.x + threadIdx.x;
    if (idx >= (size_t)total) return;
    int d = (int)(idx % D);
    size_t t = idx / D;
    int h = (int)(t % H);
    int e = (int)(t / H);
    int s = src[e], q = dst[e];
    float v = leaky(el[s * H + h] + er[q * H + h]);
    float a = expf(v - emax[q * H + h]) / denom[q * H + h];
    atomicAdd(&rst[(size_t)q * H * D + (size_t)h * D + d],
              feat[(size_t)s * H * D + (size_t)h * D + d] * a);
}

__device__ __forceinline__ float mish(float v) {
    float sp = (v > 20.f) ? v : log1pf(expf(v));
    return v * tanhf(sp);
}

// out = mish(rst + resid + bias) for hidden layers (identity residual).
__global__ void gat_finalize_mish(const float* __restrict__ rst,
                                  const float* __restrict__ resid,
                                  const float* __restrict__ bias,
                                  float* __restrict__ out, int N, int F) {
    size_t i = (size_t)blockIdx.x * blockDim.x + threadIdx.x;
    if (i >= (size_t)N * F) return;
    int f = (int)(i % F);
    out[i] = mish(rst[i] + resid[i] + bias[f]);
}

// out += res2 + bias (no activation; mean over 1 head == identity).
__global__ void gat_finalize_out(float* __restrict__ out,
                                 const float* __restrict__ res2,
                                 const float* __restrict__ bias, int N, int C) {
    int idx = blockIdx.x * blockDim.x + threadIdx.x;
    if (idx >= N * C) return;
    int c = idx % C;
    out[idx] = out[idx] + res2[idx] + bias[c];
}

// ---------------------------------------------------------------------------
extern "C" void kernel_launch(void* const* d_in, const int* in_sizes, int n_in,
                              void* d_out, int out_size, void* d_ws, size_t ws_size,
                              hipStream_t stream) {
    (void)in_sizes; (void)n_in; (void)out_size; (void)ws_size;

    const float* x     = (const float*)d_in[0];
    const int*   src   = (const int*)d_in[1];
    const int*   dst   = (const int*)d_in[2];
    const float* W0    = (const float*)d_in[3];
    const float* al0   = (const float*)d_in[4];
    const float* ar0   = (const float*)d_in[5];
    const float* b0    = (const float*)d_in[6];
    const float* W1    = (const float*)d_in[7];
    const float* al1   = (const float*)d_in[8];
    const float* ar1   = (const float*)d_in[9];
    const float* b1    = (const float*)d_in[10];
    const float* W2    = (const float*)d_in[11];
    const float* al2   = (const float*)d_in[12];
    const float* ar2   = (const float*)d_in[13];
    const float* b2    = (const float*)d_in[14];
    const float* resW2 = (const float*)d_in[15];
    float* out = (float*)d_out;

    const int N = N_NODES, E = N_EDGES, F = FDIM;

    // Workspace carve-up (floats).
    float* ws   = (float*)d_ws;
    size_t o    = 0;
    float* feat = ws + o; o += (size_t)N * F;   // projected features (current layer)
    float* rstA = ws + o; o += (size_t)N * F;   // layer-0 output
    float* rstB = ws + o; o += (size_t)N * F;   // layer-1 output
    float* el   = ws + o; o += (size_t)N * 4;
    float* er   = ws + o; o += (size_t)N * 4;
    float* emax = ws + o; o += (size_t)N * 4;
    float* denom= ws + o; o += (size_t)N * 4;
    float* res2 = ws + o; o += (size_t)N * 16;  // layer-2 residual projection

    const int TB = 256;
    const dim3 blkG(GEMM_TPB);
    const float NEG_INF = -INFINITY;
    #define NB(n) ((unsigned)(((size_t)(n) + TB - 1) / TB))
    #define GEMM_GRID(M_, N_) dim3(((M_) + 63) / 64, (N_) / 16)

    // ---------------- Layer 0: 256 -> 4 heads x 64, mish ----------------
    {
        const int H = 4, D = 64, Fout = 256;
        gat_gemm_wmma<<<GEMM_GRID(N, Fout), blkG, 0, stream>>>(x, W0, feat, N, F, Fout);
        gat_node_attn<<<NB(N * H), TB, 0, stream>>>(feat, al0, ar0, el, er, N, H, D);
        fill_f32<<<NB(N * H), TB, 0, stream>>>(emax, NEG_INF, (size_t)N * H);
        fill_f32<<<NB(N * H), TB, 0, stream>>>(denom, 0.f, (size_t)N * H);
        fill_f32<<<NB((size_t)N * Fout), TB, 0, stream>>>(rstA, 0.f, (size_t)N * Fout);
        gat_edge_max<<<NB(E * H), TB, 0, stream>>>(src, dst, el, er, emax, E, H);
        gat_edge_sum<<<NB(E * H), TB, 0, stream>>>(src, dst, el, er, emax, denom, E, H);
        long tot = (long)E * H * D;
        gat_aggregate<<<NB(tot), TB, 0, stream>>>(src, dst, feat, el, er, emax, denom, rstA, tot, H, D);
        gat_finalize_mish<<<NB((size_t)N * Fout), TB, 0, stream>>>(rstA, x, b0, rstA, N, Fout);
    }

    // ---------------- Layer 1: 256 -> 4 heads x 64, mish ----------------
    {
        const int H = 4, D = 64, Fout = 256;
        gat_gemm_wmma<<<GEMM_GRID(N, Fout), blkG, 0, stream>>>(rstA, W1, feat, N, F, Fout);
        gat_node_attn<<<NB(N * H), TB, 0, stream>>>(feat, al1, ar1, el, er, N, H, D);
        fill_f32<<<NB(N * H), TB, 0, stream>>>(emax, NEG_INF, (size_t)N * H);
        fill_f32<<<NB(N * H), TB, 0, stream>>>(denom, 0.f, (size_t)N * H);
        fill_f32<<<NB((size_t)N * Fout), TB, 0, stream>>>(rstB, 0.f, (size_t)N * Fout);
        gat_edge_max<<<NB(E * H), TB, 0, stream>>>(src, dst, el, er, emax, E, H);
        gat_edge_sum<<<NB(E * H), TB, 0, stream>>>(src, dst, el, er, emax, denom, E, H);
        long tot = (long)E * H * D;
        gat_aggregate<<<NB(tot), TB, 0, stream>>>(src, dst, feat, el, er, emax, denom, rstB, tot, H, D);
        gat_finalize_mish<<<NB((size_t)N * Fout), TB, 0, stream>>>(rstB, rstA, b1, rstB, N, Fout);
    }

    // ---------------- Layer 2: 256 -> 1 head x 16, linear residual ----------------
    {
        const int H = 1, D = 16, Fout = 16;
        gat_gemm_wmma<<<GEMM_GRID(N, Fout), blkG, 0, stream>>>(rstB, W2, feat, N, F, Fout);
        gat_gemm_wmma<<<GEMM_GRID(N, Fout), blkG, 0, stream>>>(rstB, resW2, res2, N, F, Fout);
        gat_node_attn<<<NB(N * H), TB, 0, stream>>>(feat, al2, ar2, el, er, N, H, D);
        fill_f32<<<NB(N * H), TB, 0, stream>>>(emax, NEG_INF, (size_t)N * H);
        fill_f32<<<NB(N * H), TB, 0, stream>>>(denom, 0.f, (size_t)N * H);
        fill_f32<<<NB((size_t)N * Fout), TB, 0, stream>>>(out, 0.f, (size_t)N * Fout);
        gat_edge_max<<<NB(E * H), TB, 0, stream>>>(src, dst, el, er, emax, E, H);
        gat_edge_sum<<<NB(E * H), TB, 0, stream>>>(src, dst, el, er, emax, denom, E, H);
        long tot = (long)E * H * D;
        gat_aggregate<<<NB(tot), TB, 0, stream>>>(src, dst, feat, el, er, emax, denom, out, tot, H, D);
        gat_finalize_out<<<NB(N * Fout), TB, 0, stream>>>(out, res2, b2, N, Fout);
    }
    #undef NB
    #undef GEMM_GRID
}
